// TrajectoryDecoder_79886391706022
// MI455X (gfx1250) — matrix operand order repaired and checked by reading, not compiled, not run on hardware
//
#include <hip/hip_runtime.h>

typedef __bf16 bf16_t;
typedef __attribute__((ext_vector_type(16))) __bf16 v16bf;
typedef __attribute__((ext_vector_type(8)))  float  v8f;

#define L_   3
#define H_   512
#define T_   64
#define B_   2048
#define G3H  1536   // 3*H

// ---------------------------------------------------------------------------
// Prepack: fp32 weights -> bf16 in WMMA-B operand tile layout.
// B operand of v_wmma_f32_16x16x32_bf16 is 32(K) x 16(N):
//   lane holds column N = nt*16 + lane%16
//   lanes 0-15 hold K = kt*32 + {0..7, 16..23}, lanes 16-31 hold K = kt*32 + {8..15, 24..31}
// Tile = 512 bf16 (1 KB), per-lane chunk = 16 contiguous bf16 (32 B).
// Tile order: tile = nt*ktiles + kt (K-contiguous streaming).
// transposed=1: Bmat[k,n] = src[n*Ktot + k]  (x @ W.T case)
// transposed=0: Bmat[k,n] = src[k*Ntot + n]  (x @ W   case)
// ---------------------------------------------------------------------------
__global__ void prepack_kernel(const float* __restrict__ src, bf16_t* __restrict__ dst,
                               int Ktot, int Ntot, int nbatch, int transposed) {
  size_t per = (size_t)Ktot * Ntot;
  size_t total = per * (size_t)nbatch;
  size_t idx = (size_t)blockIdx.x * blockDim.x + threadIdx.x;
  if (idx >= total) return;
  int    b    = (int)(idx / per);
  size_t r    = idx % per;
  int    tile = (int)(r >> 9);
  int    rem  = (int)(r & 511);
  int    lane = rem >> 4;
  int    e    = rem & 15;
  int ktiles  = Ktot >> 5;
  int nt = tile / ktiles;
  int kt = tile % ktiles;
  int n  = nt * 16 + (lane & 15);
  int k  = kt * 32 + ((lane < 16) ? 0 : 8) + (e & 7) + ((e & 8) << 1);
  float v = transposed ? src[b * per + (size_t)n * Ktot + k]
                       : src[b * per + (size_t)k * Ntot + n];
  dst[idx] = (bf16_t)v;
}

// ---------------------------------------------------------------------------
// Device helpers
// ---------------------------------------------------------------------------
__device__ __forceinline__ v8f zero8() {
  v8f r = {0.f, 0.f, 0.f, 0.f, 0.f, 0.f, 0.f, 0.f};
  return r;
}

__device__ __forceinline__ v8f wmma_bf16(v16bf a, v16bf b, v8f c) {
  return __builtin_amdgcn_wmma_f32_16x16x32_bf16(false, a, false, b, (short)0, c,
                                                 false, false);
}

// Prepacked B fragment: 16 contiguous bf16 per lane (2x global_load_b128)
__device__ __forceinline__ v16bf ld_bfrag(const bf16_t* base, int tile, int lane) {
  return *(const v16bf*)(base + ((size_t)tile << 9) + (lane << 4));
}

// A fragment from LDS bf16 copy already stored in A-operand order:
// layer base + kt*512 elems, per-lane 16 contiguous bf16 (2x ds_load_b128).
__device__ __forceinline__ v16bf ld_afrag_lds(const bf16_t* base, int kt, int lane) {
  return *(const v16bf*)(base + (kt << 9) + (lane << 4));
}

// Build A fragment (16x32 bf16, M=lane%16) from an fp32 row pointer (global).
// lanes 0-15 hold K = kt*32 + {0..7, 16..23}, lanes 16-31 hold {8..15, 24..31};
// caller passes p = row_base + kt*32 + (lane>>4)*8; runs are p[0..7], p[16..23].
__device__ __forceinline__ v16bf cvt_afrag(const float* p) {
  float4 x0 = *(const float4*)(p);
  float4 x1 = *(const float4*)(p + 4);
  float4 y0 = *(const float4*)(p + 16);
  float4 y1 = *(const float4*)(p + 20);
  v16bf r;
  r[0]  = (__bf16)x0.x; r[1]  = (__bf16)x0.y; r[2]  = (__bf16)x0.z; r[3]  = (__bf16)x0.w;
  r[4]  = (__bf16)x1.x; r[5]  = (__bf16)x1.y; r[6]  = (__bf16)x1.z; r[7]  = (__bf16)x1.w;
  r[8]  = (__bf16)y0.x; r[9]  = (__bf16)y0.y; r[10] = (__bf16)y0.z; r[11] = (__bf16)y0.w;
  r[12] = (__bf16)y1.x; r[13] = (__bf16)y1.y; r[14] = (__bf16)y1.z; r[15] = (__bf16)y1.w;
  return r;
}

// Scatter one h element (row m, feature k) into the bf16 A-packed LDS layout.
// Element (m,k): kt=k>>5, ko=k&31, group g=ko>>3:
//   lane' = m + 16*(g&1), elem = (ko&7) + 8*(g>=2)
__device__ __forceinline__ void store_hb(bf16_t* base, int m, int k, float val) {
  int kt = k >> 5, ko = k & 31, g = ko >> 3;
  int lanep = m + ((g & 1) << 4);
  int elem  = (ko & 7) + ((g & 2) << 2);
  base[(kt << 9) + (lanep << 4) + elem] = (bf16_t)val;
}

__device__ __forceinline__ float sigmoidf_(float x) { return 1.f / (1.f + __expf(-x)); }
__device__ __forceinline__ float tanhf_(float x) {
  float e = __expf(-2.f * fabsf(x));
  float r = (1.f - e) / (1.f + e);
  return copysignf(r, x);
}

// ---------------------------------------------------------------------------
// Main decoder: 1 workgroup = 16 batch rows, full T=64 loop in-kernel.
// 8 wave32 per WG; wave w owns hidden-unit j-tiles {w, w+8, w+16, w+24}.
// ---------------------------------------------------------------------------
__launch_bounds__(256)
__global__ void traj_decoder_kernel(
    const float* __restrict__ z, const float* __restrict__ cond,
    const bf16_t* __restrict__ hpWp, const float* __restrict__ hp_b,
    const float* __restrict__ W_ih0,
    const bf16_t* __restrict__ wihp, const bf16_t* __restrict__ whhp,
    const float* __restrict__ b_ih, const float* __restrict__ b_hh,
    const bf16_t* __restrict__ w1p, const float* __restrict__ b1,
    const float* __restrict__ W2, const float* __restrict__ b2,
    float* __restrict__ out)
{
  __shared__ float  sh_h[L_][16][H_];              // 96 KB: h state (fp32, for h_old)
  __shared__ __align__(32) bf16_t sh_hb[L_][16 * H_]; // 48 KB: h in WMMA-A packed bf16
  __shared__ float  sh_mlp[16][H_];                // 32 KB: MLP hidden
  __shared__ float  sh_pos[16][2];
  __shared__ float  sh_red[16][2][8];

  const int tid  = threadIdx.x;
  const int wave = tid >> 5;
  const int lane = tid & 31;
  const int mrow = lane & 15;
  const int half = lane >> 4;
  const int m0   = blockIdx.x * 16;

  // ---------------- h0 = [z|cond] @ hp_W.T + hp_b  (K=256, N=1536) -----------
  #pragma unroll 2
  for (int i = 0; i < 12; ++i) {
    int nt = wave + 8 * i;                 // 96 n-tiles / 8 waves
    v8f acc = zero8();
    #pragma unroll
    for (int kt = 0; kt < 8; ++kt) {
      const float* srcp = (kt < 4)
          ? (z    + (size_t)(m0 + mrow) * 128 + kt * 32 + half * 8)
          : (cond + (size_t)(m0 + mrow) * 128 + (kt - 4) * 32 + half * 8);
      v16bf a = cvt_afrag(srcp);
      v16bf b = ld_bfrag(hpWp, nt * 8 + kt, lane);
      acc = wmma_bf16(a, b, acc);
    }
    int ncol = nt * 16 + mrow;             // C layout: N = lane%16
    int l = ncol >> 9, j = ncol & 511;
    float bias = hp_b[ncol];
    #pragma unroll
    for (int v = 0; v < 8; ++v) {
      float hv = acc[v] + bias;
      sh_h[l][half * 8 + v][j] = hv;
      store_hb(&sh_hb[l][0], half * 8 + v, j, hv);
    }
  }
  if (tid < 32) sh_pos[tid >> 1][tid & 1] = 0.f;
  __syncthreads();

  // ---------------- time-step loop ------------------------------------------
  #pragma unroll 1
  for (int t = 0; t < T_; ++t) {
    // Prefetch next step's W1 tile toward the caches (no LOADcnt cost).
    if (t + 1 < T_) {
      const bf16_t* nw = w1p + (size_t)(t + 1) * (H_ * H_);
      #pragma unroll
      for (int q = 0; q < 16; ++q)
        __builtin_prefetch(nw + ((size_t)(q * 256 + tid) * 64), 0, 1);
    }

    // ----- 3 GRU layers -----
    #pragma unroll 1
    for (int l = 0; l < L_; ++l) {
      float hnew[4][8];
      const bf16_t* whh_l = whhp + (size_t)l * (96 * 16 * 512);
      const bf16_t* wih_l = wihp + (size_t)(l > 0 ? (l - 1) : 0) * (96 * 16 * 512);
      const bf16_t* hsrc  = &sh_hb[l][0];
      const bf16_t* xsrc  = (l > 0) ? &sh_hb[l - 1][0] : &sh_hb[0][0];
      #pragma unroll
      for (int i = 0; i < 4; ++i) {
        int jt = wave + 8 * i;             // hidden-unit tile (32 tiles / 8 waves)
        v8f hR = zero8(), hZ = zero8(), hN = zero8();
        v8f iR = zero8(), iZ = zero8(), iN = zero8();
        #pragma unroll 4
        for (int kt = 0; kt < 16; ++kt) {
          v16bf ah = ld_afrag_lds(hsrc, kt, lane);
          hR = wmma_bf16(ah, ld_bfrag(whh_l, ( 0 + jt) * 16 + kt, lane), hR);
          hZ = wmma_bf16(ah, ld_bfrag(whh_l, (32 + jt) * 16 + kt, lane), hZ);
          hN = wmma_bf16(ah, ld_bfrag(whh_l, (64 + jt) * 16 + kt, lane), hN);
          if (l > 0) {
            v16bf ax = ld_afrag_lds(xsrc, kt, lane);
            iR = wmma_bf16(ax, ld_bfrag(wih_l, ( 0 + jt) * 16 + kt, lane), iR);
            iZ = wmma_bf16(ax, ld_bfrag(wih_l, (32 + jt) * 16 + kt, lane), iZ);
            iN = wmma_bf16(ax, ld_bfrag(wih_l, (64 + jt) * 16 + kt, lane), iN);
          }
        }
        int nj = jt * 16 + mrow;
        float bir = b_ih[l * G3H +        nj], bhr = b_hh[l * G3H +        nj];
        float biz = b_ih[l * G3H +  512 + nj], bhz = b_hh[l * G3H +  512 + nj];
        float bin = b_ih[l * G3H + 1024 + nj], bhn = b_hh[l * G3H + 1024 + nj];
        float wr0 = 0.f, wr1 = 0.f, wz0 = 0.f, wz1 = 0.f, wn0 = 0.f, wn1 = 0.f;
        if (l == 0) {                      // layer-0 gi: K=2, do in VALU
          wr0 = W_ih0[(       nj) * 2 + 0]; wr1 = W_ih0[(       nj) * 2 + 1];
          wz0 = W_ih0[( 512 + nj) * 2 + 0]; wz1 = W_ih0[( 512 + nj) * 2 + 1];
          wn0 = W_ih0[(1024 + nj) * 2 + 0]; wn1 = W_ih0[(1024 + nj) * 2 + 1];
        }
        #pragma unroll
        for (int v = 0; v < 8; ++v) {
          int mc = half * 8 + v;
          float ir, iz, in_;
          if (l == 0) {
            float p0 = sh_pos[mc][0], p1 = sh_pos[mc][1];
            ir  = fmaf(p0, wr0, fmaf(p1, wr1, bir));
            iz  = fmaf(p0, wz0, fmaf(p1, wz1, biz));
            in_ = fmaf(p0, wn0, fmaf(p1, wn1, bin));
          } else {
            ir = iR[v] + bir; iz = iZ[v] + biz; in_ = iN[v] + bin;
          }
          float hr = hR[v] + bhr, hz = hZ[v] + bhz, hn = hN[v] + bhn;
          float r  = sigmoidf_(ir + hr);
          float zg = sigmoidf_(iz + hz);
          float n  = tanhf_(in_ + r * hn);
          float hold = sh_h[l][mc][nj];
          hnew[i][v] = (1.f - zg) * n + zg * hold;
        }
      }
      __syncthreads();                     // all reads of h[l] complete
      #pragma unroll
      for (int i = 0; i < 4; ++i) {
        int nj = (wave + 8 * i) * 16 + mrow;
        #pragma unroll
        for (int v = 0; v < 8; ++v) {
          float hv = hnew[i][v];
          sh_h[l][half * 8 + v][nj] = hv;
          store_hb(&sh_hb[l][0], half * 8 + v, nj, hv);
        }
      }
      __syncthreads();                     // hnew visible before next layer
    }

    // ----- per-step MLP layer 1: relu(h2 @ W1[t] + b1[t])  (K=512, N=512) ---
    const bf16_t* w1_t = w1p + (size_t)t * (H_ * H_);
    const bf16_t* xm   = &sh_hb[2][0];
    #pragma unroll
    for (int i = 0; i < 4; ++i) {
      int nt = wave + 8 * i;               // 32 n-tiles / 8 waves
      v8f acc = zero8();
      #pragma unroll 4
      for (int kt = 0; kt < 16; ++kt) {
        v16bf a = ld_afrag_lds(xm, kt, lane);
        acc = wmma_bf16(a, ld_bfrag(w1_t, nt * 16 + kt, lane), acc);
      }
      int nj = nt * 16 + mrow;
      float bb = b1[t * H_ + nj];
      #pragma unroll
      for (int v = 0; v < 8; ++v)
        sh_mlp[half * 8 + v][nj] = fmaxf(acc[v] + bb, 0.f);
    }
    __syncthreads();

    // ----- MLP layer 2 (512 -> 2), pos accumulate, output -------------------
    {
      int m  = tid >> 4;
      int sub = tid & 15;
      int o  = sub & 1;
      int kc = sub >> 1;                   // 8 K-chunks, stride-8 interleave
      float partial = 0.f;
      #pragma unroll 8
      for (int jj = 0; jj < 64; ++jj) {
        int k = kc + (jj << 3);
        partial = fmaf(sh_mlp[m][k], W2[(size_t)t * 1024 + k * 2 + o], partial);
      }
      sh_red[m][o][kc] = partial;
    }
    __syncthreads();
    if (tid < 32) {
      int m2 = tid >> 1, o2 = tid & 1;
      float d = b2[t * 2 + o2];
      #pragma unroll
      for (int kc = 0; kc < 8; ++kc) d += sh_red[m2][o2][kc];
      float np = sh_pos[m2][o2] + d;
      sh_pos[m2][o2] = np;
      out[(size_t)(m0 + m2) * (T_ * 2) + t * 2 + o2] = np;
    }
    __syncthreads();                       // pos visible to next step layer 0
  }
}

// ---------------------------------------------------------------------------
extern "C" void kernel_launch(void* const* d_in, const int* in_sizes, int n_in,
                              void* d_out, int out_size, void* d_ws, size_t ws_size,
                              hipStream_t stream) {
  (void)in_sizes; (void)n_in; (void)out_size; (void)ws_size;
  const float* z     = (const float*)d_in[0];
  const float* cond  = (const float*)d_in[1];
  const float* hp_W  = (const float*)d_in[2];
  const float* hp_b  = (const float*)d_in[3];
  const float* W_ih0 = (const float*)d_in[4];
  const float* W_ihr = (const float*)d_in[5];
  const float* W_hh  = (const float*)d_in[6];
  const float* b_ih  = (const float*)d_in[7];
  const float* b_hh  = (const float*)d_in[8];
  const float* W1    = (const float*)d_in[9];
  const float* b1    = (const float*)d_in[10];
  const float* W2    = (const float*)d_in[11];
  const float* b2    = (const float*)d_in[12];
  float* out = (float*)d_out;

  // Workspace layout (bf16): prepacked WMMA-B tiles, ~42 MB total (L2-resident)
  bf16_t* ws   = (bf16_t*)d_ws;
  bf16_t* hpWp = ws;
  size_t  o1   = (size_t)1536 * 256;          // hp_W
  bf16_t* whhp = ws + o1;
  size_t  o2   = o1 + (size_t)3 * 1536 * 512; // W_hh
  bf16_t* wihp = ws + o2;
  size_t  o3   = o2 + (size_t)2 * 1536 * 512; // W_ih_rest
  bf16_t* w1p  = ws + o3;                     // W1: 64 * 512*512

  auto pp = [&](const float* s, bf16_t* d, int K, int N, int nb, int tr) {
    size_t tot = (size_t)nb * K * N;
    int blocks = (int)((tot + 255) / 256);
    prepack_kernel<<<blocks, 256, 0, stream>>>(s, d, K, N, nb, tr);
  };
  pp(hp_W,  hpWp, 256, 1536,  1, 1);
  pp(W_hh,  whhp, 512, 1536,  3, 1);
  pp(W_ihr, wihp, 512, 1536,  2, 1);
  pp(W1,    w1p,  512,  512, 64, 0);

  traj_decoder_kernel<<<dim3(B_ / 16), dim3(256), 0, stream>>>(
      z, cond, hpWp, hp_b, W_ih0, wihp, whhp, b_ih, b_hh, w1p, b1, W2, b2, out);
}
